// DeltaNetMolecular_8770323218780
// MI455X (gfx1250) — compile-verified, matrix operand order checked by reading, not run on hardware
//
#include <hip/hip_runtime.h>

// ---------------------------------------------------------------------------
// CDNA5 (gfx1250) WMMA types & helpers
// ---------------------------------------------------------------------------
typedef __attribute__((ext_vector_type(16))) _Float16 v16h;
typedef __attribute__((ext_vector_type(8)))  float    v8f;

__device__ __forceinline__ v8f wmma_f16(v16h a, v16h b, v8f c) {
  // D = A(16x32 f16) * B(32x16 f16) + C(16x16 f32)
  return __builtin_amdgcn_wmma_f32_16x16x32_f16(false, a, false, b, (short)0, c,
                                                false, false);
}

__device__ __forceinline__ float silu_f(float x) {
  return x / (1.f + __expf(-x));
}

// A-fragment loader: 16x32 f16 tile from LDS, row-major stride `ld` (halfs).
// Layout per ISA 7.12.2: lanes 0-15 hold M=0..15; VGPR v holds K pair at
// k = (v&3)*2 + h*8 + (v>>2)*16  (h = lane/16).
__device__ __forceinline__ v16h frag_a(const _Float16* __restrict__ p, int ld,
                                       int lane, int klim) {
  const int h = lane >> 4;
  const _Float16* rp = p + (lane & 15) * ld;
  v16h a;
  if (klim >= 32) {
#pragma unroll
    for (int v = 0; v < 8; ++v) {
      const int k = ((v & 3) << 1) + (h << 3) + ((v >> 2) << 4);
      a[2 * v]     = rp[k];
      a[2 * v + 1] = rp[k + 1];
    }
  } else {
#pragma unroll
    for (int v = 0; v < 8; ++v) {
      const int k = ((v & 3) << 1) + (h << 3) + ((v >> 2) << 4);
      a[2 * v]     = (k < klim)     ? rp[k]     : (_Float16)0.f;
      a[2 * v + 1] = (k + 1 < klim) ? rp[k + 1] : (_Float16)0.f;
    }
  }
  return a;
}

// B fragments are pre-packed fragment-major: frag = 512 halfs, lane gets 16
// contiguous halfs -> two global_load_b128 per fragment.
__device__ __forceinline__ v16h frag_b(const _Float16* __restrict__ Wf,
                                       int fragIdx, int lane) {
  return *(const v16h*)(Wf + (size_t)fragIdx * 512 + (size_t)lane * 16);
}

// ---------------------------------------------------------------------------
// Weight packing: row-major fp32 [K x N] -> f16 WMMA-B fragment-major layout
// fragIdx = kt*Ntiles + nt ; within fragment: lane*16 + p, where
// n = nt*16 + (lane&15), k = kt*32 + kmap(p>>1, lane>>4) + (p&1)
// ---------------------------------------------------------------------------
__global__ void pack_b_frag(const float* __restrict__ W, _Float16* __restrict__ out,
                            int K, int Nn, int Ktiles, int Ntiles) {
  const int total = Ktiles * Ntiles * 512;
  for (int i = blockIdx.x * blockDim.x + threadIdx.x; i < total;
       i += gridDim.x * blockDim.x) {
    const int t = i & 511;
    const int frag = i >> 9;
    const int nt = frag % Ntiles;
    const int kt = frag / Ntiles;
    const int lane = t >> 4;
    const int p = t & 15;
    const int h = lane >> 4;
    const int n = nt * 16 + (lane & 15);
    const int v = p >> 1;
    const int k = kt * 32 + ((v & 3) << 1) + (h << 3) + ((v >> 2) << 4) + (p & 1);
    float val = (k < K && n < Nn) ? W[(size_t)k * Nn + n] : 0.f;
    out[i] = (_Float16)val;
  }
}

// ---------------------------------------------------------------------------
// Node embedding gather: featcat[:,0:128] = emb[atomids], featsh = f16 copy
// ---------------------------------------------------------------------------
__global__ void embed_kernel(const int* __restrict__ atomids,
                             const float* __restrict__ emb,
                             float* __restrict__ featcat,
                             _Float16* __restrict__ featsh, int N) {
  int i = blockIdx.x * blockDim.x + threadIdx.x;
  if (i >= N * 128) return;
  int n = i >> 7, j = i & 127;
  float v = emb[(size_t)atomids[n] * 128 + j];
  featcat[(size_t)n * 768 + j] = v;
  featsh[i] = (_Float16)v;
}

// ---------------------------------------------------------------------------
// Edge attributes: [sin(d/2^j) x32, cos x32, d, 0-pad] as f16, stride 66
// ---------------------------------------------------------------------------
__global__ void edge_attr_kernel(const float* __restrict__ coords,
                                 const int* __restrict__ ei,
                                 _Float16* __restrict__ eah, int E) {
  int e = blockIdx.x * blockDim.x + threadIdx.x;
  if (e >= E) return;
  int s = ei[e], d_ = ei[E + e];
  float dx = coords[3 * s] - coords[3 * d_];
  float dy = coords[3 * s + 1] - coords[3 * d_ + 1];
  float dz = coords[3 * s + 2] - coords[3 * d_ + 2];
  float dd = dx * dx + dy * dy + dz * dz;
  _Float16* o = eah + (size_t)e * 66;
#pragma unroll
  for (int j = 0; j < 32; ++j) {
    float x = ldexpf(dd, -j);
    o[j] = (_Float16)__sinf(x);
    o[32 + j] = (_Float16)__cosf(x);
  }
  o[64] = (_Float16)dd;
  o[65] = (_Float16)0.f;
}

__global__ void deg_kernel(const int* __restrict__ ei, float* __restrict__ deg,
                           int E) {
  int e = blockIdx.x * blockDim.x + threadIdx.x;
  if (e < E) unsafeAtomicAdd(&deg[ei[E + e]], 1.f);
}

// ---------------------------------------------------------------------------
// Edge MLP (per EGNN layer, the FLOP-dominant kernel).
// One block = 32 edges, 256 threads (8 wave32).
//  A0  [32 x 352] f16 = concat(feats[dst], feats[src], ea, pad)    (LDS)
//  L1: A0 @ W1f(352x656) + b1 -> silu -> H [32 x 656] f16          (LDS)
//  L2: H  @ W2f(672x32)  + b2 -> silu -> M2 [32 x 32] f32          (LDS, alias)
//  LN per edge row, then global_atomic_add_f32 scatter into m_sum[dst]
// ---------------------------------------------------------------------------
__global__ void __launch_bounds__(256)
edge_mlp_kernel(const _Float16* __restrict__ featsh,
                const _Float16* __restrict__ eah, const int* __restrict__ ei,
                int E, const _Float16* __restrict__ W1f,
                const _Float16* __restrict__ W2f, const float* __restrict__ b1,
                const float* __restrict__ b2, const float* __restrict__ lng,
                const float* __restrict__ lnb, float* __restrict__ m_sum) {
  __shared__ _Float16 sA[32 * 352];   // 22.5 KB (reused as M2 fp32 4 KB)
  __shared__ _Float16 sH[32 * 656];   // 41 KB
  __shared__ int sSrc[32];
  __shared__ int sDst[32];
  float* sM2 = (float*)sA;

  const int tid = threadIdx.x;
  const int w = tid >> 5, lane = tid & 31;
  const int e0 = blockIdx.x * 32;

  if (tid < 32) {
    int e = e0 + tid;
    sSrc[tid] = (e < E) ? ei[e] : 0;
    sDst[tid] = (e < E) ? ei[E + e] : 0;
  }
  __syncthreads();

  // build gathered A0 tile
  for (int i = tid; i < 32 * 352; i += 256) {
    int row = i / 352, c = i - row * 352;
    int e = e0 + row;
    _Float16 v = (_Float16)0.f;
    if (e < E) {
      if (c < 128)
        v = featsh[(size_t)sDst[row] * 128 + c];
      else if (c < 256)
        v = featsh[(size_t)sSrc[row] * 128 + (c - 128)];
      else if (c < 322)
        v = eah[(size_t)e * 66 + (c - 256)];
    }
    sA[i] = v;
  }
  __syncthreads();

  // layer 1: 41 column tiles across 8 waves, 11 k-steps, 2 row tiles
  for (int nt = w; nt < 41; nt += 8) {
    v8f acc0 = {0.f, 0.f, 0.f, 0.f, 0.f, 0.f, 0.f, 0.f};
    v8f acc1 = acc0;
    for (int kt = 0; kt < 11; ++kt) {
      v16h b = frag_b(W1f, kt * 41 + nt, lane);
      if (kt + 1 < 11)
        __builtin_prefetch(W1f + (size_t)((kt + 1) * 41 + nt) * 512 +
                               (size_t)lane * 16, 0, 3);
      v16h a0 = frag_a(sA + kt * 32, 352, lane, 352 - kt * 32);
      v16h a1 = frag_a(sA + 16 * 352 + kt * 32, 352, lane, 352 - kt * 32);
      acc0 = wmma_f16(a0, b, acc0);
      acc1 = wmma_f16(a1, b, acc1);
    }
    const int col = nt * 16 + (lane & 15);
    const float bias = (col < 642) ? b1[col] : 0.f;
    const int h = lane >> 4;
#pragma unroll
    for (int v = 0; v < 8; ++v) {
      float x0 = acc0[v] + bias;
      sH[(v + 8 * h) * 656 + col] = (_Float16)silu_f(x0);
      float x1 = acc1[v] + bias;
      sH[(16 + v + 8 * h) * 656 + col] = (_Float16)silu_f(x1);
    }
  }
  __syncthreads();

  // layer 2: 2 col tiles x 2 row tiles -> waves 0..3, 21 guarded k-steps
  if (w < 4) {
    const int nt2 = w & 1, rt = w >> 1;
    v8f acc = {0.f, 0.f, 0.f, 0.f, 0.f, 0.f, 0.f, 0.f};
    for (int kt = 0; kt < 21; ++kt) {
      v16h b = frag_b(W2f, kt * 2 + nt2, lane);
      v16h a = frag_a(sH + rt * 16 * 656 + kt * 32, 656, lane, 656 - kt * 32);
      acc = wmma_f16(a, b, acc);
    }
    const int col = nt2 * 16 + (lane & 15);
    const float bias = b2[col];
    const int h = lane >> 4;
#pragma unroll
    for (int v = 0; v < 8; ++v) {
      float x = acc[v] + bias;
      sM2[(rt * 16 + v + 8 * h) * 32 + col] = silu_f(x);
    }
  }
  __syncthreads();

  // per-edge LayerNorm + scatter-add to target node message sum
  if (tid < 32) {
    int e = e0 + tid;
    if (e < E) {
      const float* r = sM2 + tid * 32;
      float s = 0.f;
#pragma unroll
      for (int j = 0; j < 32; ++j) s += r[j];
      float mu = s * (1.f / 32.f);
      float vv = 0.f;
#pragma unroll
      for (int j = 0; j < 32; ++j) {
        float d = r[j] - mu;
        vv += d * d;
      }
      float rstd = rsqrtf(vv * (1.f / 32.f) + 1e-5f);
      float* dstp = m_sum + (size_t)sDst[tid] * 32;
#pragma unroll
      for (int j = 0; j < 32; ++j) {
        float val = (r[j] - mu) * rstd * lng[j] + lnb[j];
        unsafeAtomicAdd(dstp + j, val);
      }
    }
  }
}

// ---------------------------------------------------------------------------
// Node update MLP (per EGNN layer). One block = 32 nodes.
//  A1 [32x160] = [LN(feats) | LN(m_sum/deg)] f16
//  L1: A1 @ Wn1f(160x256) + b -> silu -> Hn [32x256] f16
//  L2: Hn @ Wn2f(256x128) + b -> O [32x128] f32 -> LN -> residual add
// ---------------------------------------------------------------------------
__global__ void __launch_bounds__(256)
node_mlp_kernel(float* __restrict__ featcat, _Float16* __restrict__ featsh,
                const float* __restrict__ m_sum, const float* __restrict__ deg,
                int N, int kidx, const _Float16* __restrict__ Wn1f,
                const _Float16* __restrict__ Wn2f, const float* __restrict__ nb1,
                const float* __restrict__ nb2, const float* __restrict__ eng,
                const float* __restrict__ enb, const float* __restrict__ n1g,
                const float* __restrict__ n1b, const float* __restrict__ n2g,
                const float* __restrict__ n2b) {
  __shared__ _Float16 sA1[32 * 160];  // 10 KB
  __shared__ _Float16 sHn[32 * 256];  // 16 KB
  __shared__ float sO[32 * 128];      // 16 KB
  const int tid = threadIdx.x, w = tid >> 5, lane = tid & 31;
  const int n0 = blockIdx.x * 32;

  if (tid < 32) {
    int n = n0 + tid;
    _Float16* a = sA1 + tid * 160;
    if (n < N) {
      const float* fx = featcat + (size_t)n * 768 + kidx * 128;
      float s = 0.f;
      for (int j = 0; j < 128; ++j) s += fx[j];
      float mu = s * (1.f / 128.f), vv = 0.f;
      for (int j = 0; j < 128; ++j) {
        float d = fx[j] - mu;
        vv += d * d;
      }
      float rstd = rsqrtf(vv * (1.f / 128.f) + 1e-5f);
      for (int j = 0; j < 128; ++j)
        a[j] = (_Float16)((fx[j] - mu) * rstd * n1g[j] + n1b[j]);

      float dg = fmaxf(deg[n], 1.f);
      const float* ms = m_sum + (size_t)n * 32;
      float t[32];
      float s2 = 0.f;
      for (int j = 0; j < 32; ++j) {
        t[j] = ms[j] / dg;
        s2 += t[j];
      }
      float mu2 = s2 * (1.f / 32.f), v2 = 0.f;
      for (int j = 0; j < 32; ++j) {
        float d = t[j] - mu2;
        v2 += d * d;
      }
      float rstd2 = rsqrtf(v2 * (1.f / 32.f) + 1e-5f);
      for (int j = 0; j < 32; ++j)
        a[128 + j] = (_Float16)((t[j] - mu2) * rstd2 * eng[j] + enb[j]);
    } else {
      for (int j = 0; j < 160; ++j) a[j] = (_Float16)0.f;
    }
  }
  __syncthreads();

  // layer 1: 16 col tiles, 5 k-steps
  for (int nt = w; nt < 16; nt += 8) {
    v8f acc0 = {0.f, 0.f, 0.f, 0.f, 0.f, 0.f, 0.f, 0.f};
    v8f acc1 = acc0;
    for (int kt = 0; kt < 5; ++kt) {
      v16h b = frag_b(Wn1f, kt * 16 + nt, lane);
      v16h a0 = frag_a(sA1 + kt * 32, 160, lane, 32);
      v16h a1 = frag_a(sA1 + 16 * 160 + kt * 32, 160, lane, 32);
      acc0 = wmma_f16(a0, b, acc0);
      acc1 = wmma_f16(a1, b, acc1);
    }
    const int col = nt * 16 + (lane & 15);
    const float bias = nb1[col];
    const int h = lane >> 4;
#pragma unroll
    for (int v = 0; v < 8; ++v) {
      float x0 = acc0[v] + bias;
      sHn[(v + 8 * h) * 256 + col] = (_Float16)silu_f(x0);
      float x1 = acc1[v] + bias;
      sHn[(16 + v + 8 * h) * 256 + col] = (_Float16)silu_f(x1);
    }
  }
  __syncthreads();

  // layer 2: 8 col tiles (one per wave), 8 k-steps, no activation
  {
    const int nt = w;
    v8f acc0 = {0.f, 0.f, 0.f, 0.f, 0.f, 0.f, 0.f, 0.f};
    v8f acc1 = acc0;
    for (int kt = 0; kt < 8; ++kt) {
      v16h b = frag_b(Wn2f, kt * 8 + nt, lane);
      v16h a0 = frag_a(sHn + kt * 32, 256, lane, 32);
      v16h a1 = frag_a(sHn + 16 * 256 + kt * 32, 256, lane, 32);
      acc0 = wmma_f16(a0, b, acc0);
      acc1 = wmma_f16(a1, b, acc1);
    }
    const int col = nt * 16 + (lane & 15);
    const float bias = nb2[col];
    const int h = lane >> 4;
#pragma unroll
    for (int v = 0; v < 8; ++v) {
      sO[(v + 8 * h) * 128 + col] = acc0[v] + bias;
      sO[(16 + v + 8 * h) * 128 + col] = acc1[v] + bias;
    }
  }
  __syncthreads();

  // LN + residual, write next feature slice (fp32 + f16)
  if (tid < 32) {
    int n = n0 + tid;
    if (n < N) {
      const float* o = sO + tid * 128;
      float s = 0.f;
      for (int j = 0; j < 128; ++j) s += o[j];
      float mu = s * (1.f / 128.f), vv = 0.f;
      for (int j = 0; j < 128; ++j) {
        float d = o[j] - mu;
        vv += d * d;
      }
      float rstd = rsqrtf(vv * (1.f / 128.f) + 1e-5f);
      const float* fold = featcat + (size_t)n * 768 + kidx * 128;
      float* fnew = featcat + (size_t)n * 768 + (kidx + 1) * 128;
      for (int j = 0; j < 128; ++j) {
        float nv = fold[j] + ((o[j] - mu) * rstd * n2g[j] + n2b[j]);
        fnew[j] = nv;
        featsh[(size_t)n * 128 + j] = (_Float16)nv;
      }
    }
  }
}

// ---------------------------------------------------------------------------
// FNN layer 0: out = silu(silu(featcat[N,768]) @ fW0 + fb0)  -> f16 [N,256]
// ---------------------------------------------------------------------------
__global__ void __launch_bounds__(256)
fnn0_kernel(const float* __restrict__ featcat, int N,
            const _Float16* __restrict__ Wf, const float* __restrict__ bias,
            _Float16* __restrict__ out) {
  __shared__ _Float16 sAf[32 * 768];  // 48 KB
  const int tid = threadIdx.x, w = tid >> 5, lane = tid & 31;
  const int n0 = blockIdx.x * 32;
  for (int i = tid; i < 32 * 768; i += 256) {
    int row = i / 768, c = i - row * 768;
    int n = n0 + row;
    float v = (n < N) ? featcat[(size_t)n * 768 + c] : 0.f;
    sAf[i] = (_Float16)silu_f(v);
  }
  __syncthreads();
  for (int nt = w; nt < 16; nt += 8) {
    v8f acc0 = {0.f, 0.f, 0.f, 0.f, 0.f, 0.f, 0.f, 0.f};
    v8f acc1 = acc0;
    for (int kt = 0; kt < 24; ++kt) {
      v16h b = frag_b(Wf, kt * 16 + nt, lane);
      v16h a0 = frag_a(sAf + kt * 32, 768, lane, 32);
      v16h a1 = frag_a(sAf + 16 * 768 + kt * 32, 768, lane, 32);
      acc0 = wmma_f16(a0, b, acc0);
      acc1 = wmma_f16(a1, b, acc1);
    }
    const int col = nt * 16 + (lane & 15);
    const float bb = bias[col];
    const int h = lane >> 4;
#pragma unroll
    for (int v = 0; v < 8; ++v) {
      int r = n0 + v + 8 * h;
      if (r < N) out[(size_t)r * 256 + col] = (_Float16)silu_f(acc0[v] + bb);
      r = n0 + 16 + v + 8 * h;
      if (r < N) out[(size_t)r * 256 + col] = (_Float16)silu_f(acc1[v] + bb);
    }
  }
}

// ---------------------------------------------------------------------------
// FNN middle/out layers: out = silu(in[N,256] @ W + b) -> f16 [N,256]
// ---------------------------------------------------------------------------
__global__ void __launch_bounds__(256)
fnn_mid_kernel(const _Float16* __restrict__ in, int N,
               const _Float16* __restrict__ Wf, const float* __restrict__ bias,
               _Float16* __restrict__ out) {
  __shared__ _Float16 sIn[32 * 256];
  const int tid = threadIdx.x, w = tid >> 5, lane = tid & 31;
  const int n0 = blockIdx.x * 32;
  for (int i = tid; i < 32 * 256; i += 256) {
    int row = i >> 8, c = i & 255;
    int n = n0 + row;
    sIn[i] = (n < N) ? in[(size_t)n * 256 + c] : (_Float16)0.f;
  }
  __syncthreads();
  for (int nt = w; nt < 16; nt += 8) {
    v8f acc0 = {0.f, 0.f, 0.f, 0.f, 0.f, 0.f, 0.f, 0.f};
    v8f acc1 = acc0;
    for (int kt = 0; kt < 8; ++kt) {
      v16h b = frag_b(Wf, kt * 16 + nt, lane);
      v16h a0 = frag_a(sIn + kt * 32, 256, lane, 32);
      v16h a1 = frag_a(sIn + 16 * 256 + kt * 32, 256, lane, 32);
      acc0 = wmma_f16(a0, b, acc0);
      acc1 = wmma_f16(a1, b, acc1);
    }
    const int col = nt * 16 + (lane & 15);
    const float bb = bias[col];
    const int h = lane >> 4;
#pragma unroll
    for (int v = 0; v < 8; ++v) {
      int r = n0 + v + 8 * h;
      if (r < N) out[(size_t)r * 256 + col] = (_Float16)silu_f(acc0[v] + bb);
      r = n0 + 16 + v + 8 * h;
      if (r < N) out[(size_t)r * 256 + col] = (_Float16)silu_f(acc1[v] + bb);
    }
  }
}

// ---------------------------------------------------------------------------
// Graph pooling
// ---------------------------------------------------------------------------
__global__ void seg_sum_kernel(const _Float16* __restrict__ f,
                               const int* __restrict__ batch, int N,
                               float* __restrict__ gsum) {
  int i = blockIdx.x * blockDim.x + threadIdx.x;
  if (i < N * 256) {
    int n = i >> 8, j = i & 255;
    unsafeAtomicAdd(&gsum[(size_t)batch[n] * 256 + j], (float)f[i]);
  }
}
__global__ void seg_cnt_kernel(const int* __restrict__ batch, int N,
                               float* __restrict__ gcnt) {
  int n = blockIdx.x * blockDim.x + threadIdx.x;
  if (n < N) unsafeAtomicAdd(&gcnt[batch[n]], 1.f);
}

// ---------------------------------------------------------------------------
// Head: g = gsum/cnt ; 3-layer MLP 256->256->256->1, fused in one block pass
// ---------------------------------------------------------------------------
__global__ void __launch_bounds__(256)
head_kernel(const float* __restrict__ gsum, const float* __restrict__ gcnt,
            int B, const _Float16* __restrict__ W0f, const float* __restrict__ b0,
            const _Float16* __restrict__ W1f, const float* __restrict__ b1v,
            const float* __restrict__ W2, const float* __restrict__ b2v,
            float* __restrict__ out) {
  __shared__ _Float16 sG[32 * 256];
  __shared__ _Float16 sT[32 * 256];
  const int tid = threadIdx.x, w = tid >> 5, lane = tid & 31;
  const int r0 = blockIdx.x * 32;

  if (tid < 32) {
    int b = r0 + tid;
    _Float16* g = sG + tid * 256;
    if (b < B) {
      float c = fmaxf(gcnt[b], 1.f);
      for (int j = 0; j < 256; ++j) g[j] = (_Float16)(gsum[(size_t)b * 256 + j] / c);
    } else {
      for (int j = 0; j < 256; ++j) g[j] = (_Float16)0.f;
    }
  }
  __syncthreads();

  // g0: sG @ W0 -> silu -> sT
  for (int nt = w; nt < 16; nt += 8) {
    v8f acc0 = {0.f, 0.f, 0.f, 0.f, 0.f, 0.f, 0.f, 0.f};
    v8f acc1 = acc0;
    for (int kt = 0; kt < 8; ++kt) {
      v16h b = frag_b(W0f, kt * 16 + nt, lane);
      v16h a0 = frag_a(sG + kt * 32, 256, lane, 32);
      v16h a1 = frag_a(sG + 16 * 256 + kt * 32, 256, lane, 32);
      acc0 = wmma_f16(a0, b, acc0);
      acc1 = wmma_f16(a1, b, acc1);
    }
    const int col = nt * 16 + (lane & 15);
    const float bb = b0[col];
    const int h = lane >> 4;
#pragma unroll
    for (int v = 0; v < 8; ++v) {
      sT[(v + 8 * h) * 256 + col] = (_Float16)silu_f(acc0[v] + bb);
      sT[(16 + v + 8 * h) * 256 + col] = (_Float16)silu_f(acc1[v] + bb);
    }
  }
  __syncthreads();

  // g1: sT @ W1 -> silu -> sG
  for (int nt = w; nt < 16; nt += 8) {
    v8f acc0 = {0.f, 0.f, 0.f, 0.f, 0.f, 0.f, 0.f, 0.f};
    v8f acc1 = acc0;
    for (int kt = 0; kt < 8; ++kt) {
      v16h b = frag_b(W1f, kt * 16 + nt, lane);
      v16h a0 = frag_a(sT + kt * 32, 256, lane, 32);
      v16h a1 = frag_a(sT + 16 * 256 + kt * 32, 256, lane, 32);
      acc0 = wmma_f16(a0, b, acc0);
      acc1 = wmma_f16(a1, b, acc1);
    }
    const int col = nt * 16 + (lane & 15);
    const float bb = b1v[col];
    const int h = lane >> 4;
#pragma unroll
    for (int v = 0; v < 8; ++v) {
      sG[(v + 8 * h) * 256 + col] = (_Float16)silu_f(acc0[v] + bb);
      sG[(16 + v + 8 * h) * 256 + col] = (_Float16)silu_f(acc1[v] + bb);
    }
  }
  __syncthreads();

  // g2: dot with gW2[256,1]
  if (tid < 32) {
    int b = r0 + tid;
    if (b < B) {
      const _Float16* g = sG + tid * 256;
      float s = b2v[0];
      for (int j = 0; j < 256; ++j) s += (float)g[j] * W2[j];
      out[b] = s;
    }
  }
}

// ---------------------------------------------------------------------------
// Host orchestration
// ---------------------------------------------------------------------------
extern "C" void kernel_launch(void* const* d_in, const int* in_sizes, int n_in,
                              void* d_out, int out_size, void* d_ws,
                              size_t ws_size, hipStream_t stream) {
  const float* coords = (const float*)d_in[0];
  const int* atomids = (const int*)d_in[1];
  const int* ei = (const int*)d_in[2];
  const int* batch = (const int*)d_in[3];
  const float* emb = (const float*)d_in[5];
  const float* keW1 = (const float*)d_in[6];
  const float* keb1 = (const float*)d_in[7];
  const float* keW2 = (const float*)d_in[8];
  const float* keb2 = (const float*)d_in[9];
  const float* keng = (const float*)d_in[10];
  const float* kenb = (const float*)d_in[11];
  const float* kn1g = (const float*)d_in[12];
  const float* kn1b = (const float*)d_in[13];
  const float* knW1 = (const float*)d_in[14];
  const float* knb1 = (const float*)d_in[15];
  const float* knW2 = (const float*)d_in[16];
  const float* knb2 = (const float*)d_in[17];
  const float* kn2g = (const float*)d_in[18];
  const float* kn2b = (const float*)d_in[19];
  const float* fW0 = (const float*)d_in[20];
  const float* fb0 = (const float*)d_in[21];
  const float* fW1 = (const float*)d_in[22];
  const float* fb1 = (const float*)d_in[23];
  const float* fW2 = (const float*)d_in[24];
  const float* fb2 = (const float*)d_in[25];
  const float* gW0 = (const float*)d_in[26];
  const float* gb0 = (const float*)d_in[27];
  const float* gW1 = (const float*)d_in[28];
  const float* gb1 = (const float*)d_in[29];
  const float* gW2 = (const float*)d_in[30];
  const float* gb2 = (const float*)d_in[31];

  const int N = in_sizes[0] / 3;
  const int E = in_sizes[2] / 2;
  const int B = out_size;  // output is [B,1] fp32
  const int K = 5;

  // workspace partition (256-byte aligned bump allocator)
  size_t off = 0;
  auto alloc = [&](size_t bytes) -> void* {
    void* p = (char*)d_ws + off;
    off = (off + bytes + 255) & ~(size_t)255;
    return p;
  };
  float* featcat = (float*)alloc((size_t)N * 768 * 4);
  _Float16* featsh = (_Float16*)alloc((size_t)N * 128 * 2);
  _Float16* eah = (_Float16*)alloc((size_t)E * 66 * 2);
  float* deg = (float*)alloc((size_t)N * 4);
  float* m_sum = (float*)alloc((size_t)N * 32 * 4);
  _Float16* hA = (_Float16*)alloc((size_t)N * 256 * 2);
  _Float16* hB = (_Float16*)alloc((size_t)N * 256 * 2);
  float* gsum = (float*)alloc((size_t)B * 256 * 4);
  float* gcnt = (float*)alloc((size_t)B * 4);
  const size_t W1SZ = 11 * 41 * 512, W2SZ = 21 * 2 * 512;
  const size_t WN1SZ = 5 * 16 * 512, WN2SZ = 8 * 8 * 512;
  _Float16* W1h = (_Float16*)alloc(K * W1SZ * 2);
  _Float16* W2h = (_Float16*)alloc(K * W2SZ * 2);
  _Float16* Wn1h = (_Float16*)alloc(K * WN1SZ * 2);
  _Float16* Wn2h = (_Float16*)alloc(K * WN2SZ * 2);
  _Float16* fW0h = (_Float16*)alloc((size_t)24 * 16 * 512 * 2);
  _Float16* fW1h = (_Float16*)alloc((size_t)8 * 16 * 512 * 2);
  _Float16* fW2h = (_Float16*)alloc((size_t)8 * 16 * 512 * 2);
  _Float16* gW0h = (_Float16*)alloc((size_t)8 * 16 * 512 * 2);
  _Float16* gW1h = (_Float16*)alloc((size_t)8 * 16 * 512 * 2);
  (void)ws_size;
  (void)n_in;

  auto pack = [&](const float* W, _Float16* dst, int Kdim, int Nn, int Kt, int Nt) {
    int total = Kt * Nt * 512;
    pack_b_frag<<<(total + 255) / 256, 256, 0, stream>>>(W, dst, Kdim, Nn, Kt, Nt);
  };
  for (int k = 0; k < K; ++k) {
    pack(keW1 + (size_t)k * 321 * 642, W1h + k * W1SZ, 321, 642, 11, 41);
    pack(keW2 + (size_t)k * 642 * 32, W2h + k * W2SZ, 642, 32, 21, 2);
    pack(knW1 + (size_t)k * 160 * 256, Wn1h + k * WN1SZ, 160, 256, 5, 16);
    pack(knW2 + (size_t)k * 256 * 128, Wn2h + k * WN2SZ, 256, 128, 8, 8);
  }
  pack(fW0, fW0h, 768, 256, 24, 16);
  pack(fW1, fW1h, 256, 256, 8, 16);
  pack(fW2, fW2h, 256, 256, 8, 16);
  pack(gW0, gW0h, 256, 256, 8, 16);
  pack(gW1, gW1h, 256, 256, 8, 16);

  embed_kernel<<<(N * 128 + 255) / 256, 256, 0, stream>>>(atomids, emb, featcat,
                                                          featsh, N);
  edge_attr_kernel<<<(E + 255) / 256, 256, 0, stream>>>(coords, ei, eah, E);
  hipMemsetAsync(deg, 0, (size_t)N * 4, stream);
  deg_kernel<<<(E + 255) / 256, 256, 0, stream>>>(ei, deg, E);

  for (int k = 0; k < K; ++k) {
    hipMemsetAsync(m_sum, 0, (size_t)N * 32 * 4, stream);
    edge_mlp_kernel<<<(E + 31) / 32, 256, 0, stream>>>(
        featsh, eah, ei, E, W1h + k * W1SZ, W2h + k * W2SZ, keb1 + k * 642,
        keb2 + k * 32, keng + k * 32, kenb + k * 32, m_sum);
    node_mlp_kernel<<<(N + 31) / 32, 256, 0, stream>>>(
        featcat, featsh, m_sum, deg, N, k, Wn1h + k * WN1SZ, Wn2h + k * WN2SZ,
        knb1 + k * 256, knb2 + k * 128, keng + k * 32, kenb + k * 32,
        kn1g + k * 128, kn1b + k * 128, kn2g + k * 128, kn2b + k * 128);
  }

  fnn0_kernel<<<(N + 31) / 32, 256, 0, stream>>>(featcat, N, fW0h, fb0, hA);
  fnn_mid_kernel<<<(N + 31) / 32, 256, 0, stream>>>(hA, N, fW1h, fb1, hB);
  fnn_mid_kernel<<<(N + 31) / 32, 256, 0, stream>>>(hB, N, fW2h, fb2, hA);

  hipMemsetAsync(gsum, 0, (size_t)B * 256 * 4, stream);
  hipMemsetAsync(gcnt, 0, (size_t)B * 4, stream);
  seg_sum_kernel<<<(N * 256 + 255) / 256, 256, 0, stream>>>(hA, batch, N, gsum);
  seg_cnt_kernel<<<(N + 255) / 256, 256, 0, stream>>>(batch, N, gcnt);

  head_kernel<<<(B + 31) / 32, 256, 0, stream>>>(gsum, gcnt, B, gW0h, gb0, gW1h,
                                                 gb1, gW2, gb2, (float*)d_out);
}
